// IndivBlur8_58377195487762
// MI455X (gfx1250) — matrix operand order repaired and checked by reading, not compiled, not run on hardware
//
#include <hip/hip_runtime.h>

// ---------------------------------------------------------------------------
// CDNA5 (gfx1250) IndivBlur density head.
// Convs 2..5: implicit-GEMM on v_wmma_f32_16x16x32_f16, wave32.
// Im2col tiles staged block-cooperatively into LDS with CDNA5 async copies
// (global_load_async_to_lds_b128 + s_wait_asynccnt), consumed as ds_load_b128
// fragments by all 8 waves of the block.
// ---------------------------------------------------------------------------

typedef __attribute__((ext_vector_type(16))) _Float16 v16h;
typedef __attribute__((ext_vector_type(8)))  _Float16 v8h;
typedef __attribute__((ext_vector_type(8)))  float    v8f;

union HFrag { v16h v; v8h h[2]; _Float16 e[16]; };
union CFrag { v8f  v; float e[8]; };

__device__ __forceinline__ float lrelu(float x) { return x >= 0.f ? x : 0.01f * x; }

// CDNA5 async global->LDS 16B copy. LDS aperture: generic addr[31:0] is the
// LDS byte offset (ISA 10.2), so truncating the shared pointer is exact.
__device__ __forceinline__ void async_copy_b128(_Float16* lds, const _Float16* g) {
  unsigned lds_off = (unsigned)(unsigned long long)lds;
  asm volatile("global_load_async_to_lds_b128 %0, %1, off"
               :: "v"(lds_off), "v"(g) : "memory");
}
__device__ __forceinline__ void wait_async0() {
  asm volatile("s_wait_asynccnt 0x0" ::: "memory");
}

// --------------------- weight repack: OIHW f32 -> [Opad][9][Cin] f16 --------
__global__ void wpack_kernel(const float* __restrict__ w, _Float16* __restrict__ wf,
                             int Cin, int Opad, int Oreal) {
  int idx = blockIdx.x * blockDim.x + threadIdx.x;
  int total = Opad * 9 * Cin;
  if (idx >= total) return;
  int o   = idx / (9 * Cin);
  int rem = idx - o * 9 * Cin;
  int rs  = rem / Cin;
  int c   = rem - rs * Cin;
  float v = (o < Oreal) ? w[((long)o * Cin + c) * 9 + rs] : 0.f;
  wf[idx] = (_Float16)v;
}

// --------------------------- conv1 + lrelu + pool ---------------------------
// Cin=3 (K=27): scalar path. One thread per pooled pixel computes all 32 ch.
// img [3,1024,1024] f32 -> out NHWC f16 [512,512,32]
__global__ void conv1_pool_kernel(const float* __restrict__ img,
                                  const float* __restrict__ w,
                                  const float* __restrict__ bias,
                                  _Float16* __restrict__ out) {
  int pix = blockIdx.x * blockDim.x + threadIdx.x;
  if (pix >= 512 * 512) return;
  int yp = pix >> 9;
  int xp = pix & 511;

  float im[3][4][4];
  #pragma unroll
  for (int ci = 0; ci < 3; ++ci)
    #pragma unroll
    for (int r = 0; r < 4; ++r) {
      int iy = 2 * yp - 1 + r;
      #pragma unroll
      for (int s = 0; s < 4; ++s) {
        int ix = 2 * xp - 1 + s;
        float v = 0.f;
        if (iy >= 0 && iy < 1024 && ix >= 0 && ix < 1024)
          v = img[(ci << 20) + (iy << 10) + ix];
        im[ci][r][s] = v;
      }
    }

  for (int c = 0; c < 32; ++c) {          // weight index uniform -> s_loads
    const float* wc = w + c * 27;
    float bb = bias[c];
    float m = -3.4e38f;
    #pragma unroll
    for (int dy = 0; dy < 2; ++dy)
      #pragma unroll
      for (int dx = 0; dx < 2; ++dx) {
        float acc = bb;
        #pragma unroll
        for (int ci = 0; ci < 3; ++ci)
          #pragma unroll
          for (int r = 0; r < 3; ++r)
            #pragma unroll
            for (int s = 0; s < 3; ++s)
              acc = fmaf(wc[ci * 9 + r * 3 + s], im[ci][dy + r][dx + s], acc);
        m = fmaxf(m, lrelu(acc));
      }
    out[(long)pix * 32 + c] = (_Float16)m;
  }
}

// --------------------------- WMMA implicit-GEMM conv ------------------------
// Block (8 waves) = PT*64 pixels x OT*16 out-channels, PT*OT == 8.
// Per 32-wide K-chunk the block stages the [PT*64 pixels x 32] activation
// slab in LDS once (async copies); each wave reads its B fragments from LDS.
template <int PT, int OT>
__global__ void __launch_bounds__(256)
conv_wmma_lds_kernel(const _Float16* __restrict__ in,   // NHWC, stride Cin
                     const _Float16* __restrict__ wf,   // [Opad][9][Cin]
                     const float* __restrict__ bias,
                     _Float16* __restrict__ out,        // NHWC, stride Cstride
                     int Cin, int H, int W,
                     int octiles, int CoutReal, int Cstride,
                     int doLrelu) {
  constexpr int PIX  = PT * 64;
  constexpr int TSTR = 40;                    // halves per pixel row (80B): banks
  __shared__ _Float16 tile[PIX * TSTR];       // walk (ln*20)%64 -> conflict-free

  const int octBlocks = (octiles + OT - 1) / OT;
  const int pbase = (blockIdx.x / octBlocks) * PIX;   // flat pixel base (row-contig)
  const int octB  = blockIdx.x % octBlocks;

  const int tid   = threadIdx.x;
  const int wslot = tid >> 5;
  const int lane  = tid & 31;
  const int hl    = lane >> 4;                // K-half select (ISA layout)
  const int ln    = lane & 15;
  const int ptSub = wslot / OT;               // which 64-pixel group of the block
  const int oct   = octB * OT + (wslot % OT);
  const bool octok = oct < octiles;           // wave-uniform

  const int y  = pbase / W;                   // PIX consecutive pixels, one row
  const int x0 = pbase - y * W;

  const int ocA = (octok ? oct : 0) * 16 + ln;        // A row this lane supplies
  const _Float16* wfrow = wf + (long)ocA * 9 * Cin;   // K-contiguous

  const v8h hz = {};
  const v8f fz = {};
  v8f acc[4] = {fz, fz, fz, fz};

  const int K = Cin * 9;
  for (int kb = 0; kb < K; kb += 32) {
    // chunk lies inside one (r,s) block: Cin is a multiple of 32
    int rs = kb / Cin;
    int cb = kb - rs * Cin;
    int r  = rs / 3;
    int s  = rs - r * 3;
    int iy = y + r - 1;

    __syncthreads();                          // previous tile fully consumed
    // cooperative stage: PIX pixels x 32 halves; thread -> 16B segment(s)
    #pragma unroll
    for (int i = 0; i < PT; ++i) {
      int segIdx = tid + i * 256;
      int pix = segIdx >> 2;                  // 0..PIX-1
      int seg = segIdx & 3;                   // 8-half segment
      int ix  = x0 + pix + s - 1;
      _Float16* dst = tile + pix * TSTR + seg * 8;
      if (iy >= 0 && iy < H && ix >= 0 && ix < W) {
        async_copy_b128(dst, in + ((long)iy * W + ix) * Cin + cb + seg * 8);
      } else {
        *(v8h*)dst = hz;                      // halo: ds_store zeros
      }
    }
    wait_async0();                            // this wave's async copies landed
    __syncthreads();                          // all waves' copies visible

    // A fragment: element i<8  <-> c = cb + hl*8 + i ; i>=8 <-> +16
    HFrag a;
    const _Float16* abase = wfrow + kb;
    a.h[0] = *(const v8h*)(abase + hl * 8);
    a.h[1] = *(const v8h*)(abase + 16 + hl * 8);
    if (kb + 32 < K)
      __builtin_prefetch(abase + 32 + hl * 8, 0, 1);  // next chunk's weights

    if (octok) {                              // wave-uniform: EXEC all-1 at WMMA
      #pragma unroll
      for (int nt = 0; nt < 4; ++nt) {
        int pixL = ptSub * 64 + nt * 16 + ln;
        // B fragment: element i<8 <-> c = cb + hl*8 + i ; i>=8 <-> +16
        HFrag b;
        const v8h* p = (const v8h*)(tile + pixL * TSTR + hl * 8);
        b.h[0] = p[0];                        // ds_load_b128 x2
        b.h[1] = p[2];
        acc[nt] = __builtin_amdgcn_wmma_f32_16x16x32_f16(
            false, a.v, false, b.v, (short)0, acc[nt], false, false);
      }
    }
  }

  if (!octok) return;
  // C/D layout: lane ln, element j <-> (M = j + 8*hl, N = ln)
  const int ocbase = oct * 16 + hl * 8;       // 8 contiguous out-channels
  #pragma unroll
  for (int nt = 0; nt < 4; ++nt) {
    CFrag c; c.v = acc[nt];
    long pixid  = pbase + ptSub * 64 + nt * 16 + ln;
    long pixoff = pixid * Cstride + ocbase;
    if (ocbase + 8 <= CoutReal) {
      union { v8h v; _Float16 e[8]; } st;
      #pragma unroll
      for (int j = 0; j < 8; ++j) {
        float v = c.e[j] + bias[ocbase + j];
        if (doLrelu) v = lrelu(v);
        st.e[j] = (_Float16)v;
      }
      *(v8h*)(out + pixoff) = st.v;           // global_store_b128
    } else {
      #pragma unroll
      for (int j = 0; j < 8; ++j) {
        int oc = ocbase + j;
        if (oc < CoutReal) {
          float v = c.e[j] + bias[oc];
          if (doLrelu) v = lrelu(v);
          out[pixoff + j] = (_Float16)v;
        }
      }
    }
  }
}

// ------------------------------- 2x2 maxpool (NHWC f16) ---------------------
__global__ void maxpool_kernel(const _Float16* __restrict__ in,
                               _Float16* __restrict__ out,
                               int C, int Ho, int Wo) {
  int idx = blockIdx.x * blockDim.x + threadIdx.x;
  int total = C * Ho * Wo;
  if (idx >= total) return;
  int c   = idx % C;
  int pix = idx / C;
  int yp  = pix / Wo;
  int xp  = pix - yp * Wo;
  int Wi  = Wo * 2;
  const _Float16* p = in + ((long)(2 * yp) * Wi + 2 * xp) * C + c;
  long rowstep = (long)Wi * C;
  float m = fmaxf(fmaxf((float)p[0], (float)p[C]),
                  fmaxf((float)p[rowstep], (float)p[rowstep + C]));
  out[idx] = (_Float16)m;
}

// ------------------- channel shift-to-min + normalize ----------------------
// c5: NHWC f16 [4096][240] (225 valid) -> kn: CHW f32 [225][4096]
__global__ void knorm_kernel(const _Float16* __restrict__ c5,
                             float* __restrict__ kn) {
  int pos = blockIdx.x * blockDim.x + threadIdx.x;
  if (pos >= 4096) return;
  const _Float16* p = c5 + (long)pos * 240;
  float mn = 3.4e38f;
  for (int ch = 0; ch < 225; ++ch) mn = fminf(mn, (float)p[ch]);
  float s = 0.f;
  for (int ch = 0; ch < 225; ++ch) s += (float)p[ch] - mn;
  float inv = 1.f / s;
  for (int ch = 0; ch < 225; ++ch)
    kn[ch * 4096 + pos] = ((float)p[ch] - mn) * inv;
}

// ------------------------------- scatter path -------------------------------
__global__ void zero_kernel(float* __restrict__ p, int n) {
  int i = blockIdx.x * blockDim.x + threadIdx.x;
  if (i < n) p[i] = 0.f;
}

__global__ void scatter_kernel(const int* __restrict__ pts,
                               const float* __restrict__ kn,
                               float* __restrict__ dens) {
  int p = blockIdx.x;
  int t = threadIdx.x;
  if (t >= 225) return;
  int b  = p >> 11;                       // 2048 points per batch
  int px = pts[2 * p + 0];
  int py = pts[2 * p + 1];
  int ky = min(63, py >> 4);
  int kx = min(63, px >> 4);
  int y0 = max(0, (py >> 3) - 8);         // (S+1)//2 = 8
  int x0 = max(0, (px >> 3) - 8);
  int i = t / 15;
  int j = t - i * 15;
  float v = kn[t * 4096 + ky * 64 + kx];
  atomicAdd(&dens[((long)b * 143 + (y0 + i)) * 143 + (x0 + j)], v);
}

__global__ void crop_kernel(const float* __restrict__ dens,
                            float* __restrict__ out) {
  int idx = blockIdx.x * blockDim.x + threadIdx.x;
  if (idx >= 4 * 128 * 128) return;
  int b   = idx >> 14;
  int rem = idx & 16383;
  int y   = rem >> 7;
  int x   = rem & 127;
  out[idx] = dens[((long)b * 143 + y) * 143 + x];
}

// ------------------------------- launcher -----------------------------------
static inline int cdiv(long a, long b) { return (int)((a + b - 1) / b); }

extern "C" void kernel_launch(void* const* d_in, const int* in_sizes, int n_in,
                              void* d_out, int out_size, void* d_ws, size_t ws_size,
                              hipStream_t stream) {
  const float* w1 = (const float*)d_in[0];
  const float* b1 = (const float*)d_in[1];
  const float* w2 = (const float*)d_in[2];
  const float* b2 = (const float*)d_in[3];
  const float* w3 = (const float*)d_in[4];
  const float* b3 = (const float*)d_in[5];
  const float* w4 = (const float*)d_in[6];
  const float* b4 = (const float*)d_in[7];
  const float* w5 = (const float*)d_in[8];
  const float* b5 = (const float*)d_in[9];
  const int*   pts = (const int*)d_in[10];
  const float* img = (const float*)d_in[11];
  // d_in[12] = shape (constant [4,1,128,128]) — unused.

  // Workspace (peak ~51.4 MB). Activations f16; ping-pong, conv outputs
  // staggered past their inputs. Repacked weights live at 48 MB+.
  char* ws = (char*)d_ws;
  _Float16* P1 = (_Float16*)(ws + 0);                    // [512*512][32]  16 MB
  _Float16* C2 = (_Float16*)(ws + (size_t)16777216);     // [512*512][64]  32 MB
  _Float16* P2 = (_Float16*)(ws + 0);                    // [256*256][64]   8 MB
  _Float16* C3 = (_Float16*)(ws + (size_t)8388608);      // [256*256][128] 16 MB
  _Float16* P3 = (_Float16*)(ws + 0);                    // [128*128][128]  4 MB
  _Float16* C4 = (_Float16*)(ws + (size_t)4194304);      // [128*128][128]  4 MB
  _Float16* P4 = (_Float16*)(ws + 0);                    // [64*64][128]    1 MB
  _Float16* C5 = (_Float16*)(ws + (size_t)1048576);      // [64*64][240]
  float*    KN   = (float*)(ws + (size_t)4194304);       // [225][4096] f32
  float*    DENS = (float*)(ws + (size_t)8388608);       // [4][143][143]
  _Float16* WF2 = (_Float16*)(ws + (size_t)50331648);    //  64*9*32
  _Float16* WF3 = (_Float16*)(ws + (size_t)50331648 + 36864);
  _Float16* WF4 = (_Float16*)(ws + (size_t)50331648 + 184320);
  _Float16* WF5 = (_Float16*)(ws + (size_t)50331648 + 479232);

  // weight repacks (f32 OIHW -> f16 [Opad][9][Cin])
  wpack_kernel<<<cdiv(64L * 9 * 32, 256), 256, 0, stream>>>(w2, WF2, 32, 64, 64);
  wpack_kernel<<<cdiv(128L * 9 * 64, 256), 256, 0, stream>>>(w3, WF3, 64, 128, 128);
  wpack_kernel<<<cdiv(128L * 9 * 128, 256), 256, 0, stream>>>(w4, WF4, 128, 128, 128);
  wpack_kernel<<<cdiv(240L * 9 * 128, 256), 256, 0, stream>>>(w5, WF5, 128, 240, 225);

  // L1: scalar conv+lrelu+pool, one thread per pooled pixel
  conv1_pool_kernel<<<cdiv(512L * 512, 256), 256, 0, stream>>>(img, w1, b1, P1);

  // L2: 32->64 @ 512^2, K=288; block = 128 px x 64 ch (PT=2, OT=4)
  {
    int blocks = (512 * 512 / 128) * 1;               // octBlocks = 1
    conv_wmma_lds_kernel<2, 4><<<blocks, 256, 0, stream>>>(
        P1, WF2, b2, C2, 32, 512, 512, 4, 64, 64, 1);
  }
  maxpool_kernel<<<cdiv(64L * 256 * 256, 256), 256, 0, stream>>>(C2, P2, 64, 256, 256);

  // L3: 64->128 @ 256^2, K=576; block = 64 px x 128 ch (PT=1, OT=8)
  {
    int blocks = (256 * 256 / 64) * 1;
    conv_wmma_lds_kernel<1, 8><<<blocks, 256, 0, stream>>>(
        P2, WF3, b3, C3, 64, 256, 256, 8, 128, 128, 1);
  }
  maxpool_kernel<<<cdiv(128L * 128 * 128, 256), 256, 0, stream>>>(C3, P3, 128, 128, 128);

  // L4: 128->128 @ 128^2, K=1152
  {
    int blocks = (128 * 128 / 64) * 1;
    conv_wmma_lds_kernel<1, 8><<<blocks, 256, 0, stream>>>(
        P3, WF4, b4, C4, 128, 128, 128, 8, 128, 128, 1);
  }
  maxpool_kernel<<<cdiv(128L * 64 * 64, 256), 256, 0, stream>>>(C4, P4, 128, 64, 64);

  // L5: 128->225 @ 64^2 (15 oc-tiles -> 2 oct-blocks), no lrelu/pool
  {
    int blocks = (64 * 64 / 64) * 2;                  // octBlocks = 2
    conv_wmma_lds_kernel<1, 8><<<blocks, 256, 0, stream>>>(
        P4, WF5, b5, C5, 128, 64, 64, 15, 225, 240, 0);
  }

  knorm_kernel<<<16, 256, 0, stream>>>(C5, KN);

  zero_kernel<<<cdiv(4L * 143 * 143, 256), 256, 0, stream>>>(DENS, 4 * 143 * 143);
  scatter_kernel<<<4 * 2048, 256, 0, stream>>>(pts, KN, DENS);
  crop_kernel<<<cdiv(4L * 128 * 128, 256), 256, 0, stream>>>(DENS, (float*)d_out);
}
